// DFNCell_53386443489795
// MI455X (gfx1250) — matrix operand order repaired and checked by reading, not compiled, not run on hardware
//
#include <hip/hip_runtime.h>
#include <hip/hip_bf16.h>

typedef __attribute__((ext_vector_type(16))) _Float16 v16h;
typedef __attribute__((ext_vector_type(8)))  float    v8f;

__device__ __forceinline__ float lrelu(float x) { return x >= 0.f ? x : 0.3f * x; }

// Adapts any pointer to whatever pointer type a builtin's prototype demands
// (handles plain / vector / address-space-qualified pointee types).
struct AnyDevPtr {
  uintptr_t v;
  __device__ explicit AnyDevPtr(const void* p) : v((uintptr_t)p) {}
  template <typename T> __device__ operator T*() const { return (T*)v; }
};

// Cooperative copy of packed-B weights global->LDS. Uses the gfx1250 async
// LDS-DMA path (GLOBAL_LOAD_ASYNC_TO_LDS_B128, ASYNCcnt) when the toolchain
// exposes it; otherwise a plain vector copy. nChunks = 16-byte chunks.
__device__ __forceinline__ void stage_B_to_lds(const _Float16* __restrict__ wB,
                                               _Float16* smemB, int nChunks) {
#if __has_builtin(__builtin_amdgcn_global_load_async_to_lds_b128)
  for (int c = threadIdx.x; c < nChunks; c += blockDim.x) {
    __builtin_amdgcn_global_load_async_to_lds_b128(AnyDevPtr(wB + c * 8),
                                                   AnyDevPtr(smemB + c * 8), 0, 0);
  }
#if __has_builtin(__builtin_amdgcn_s_wait_asynccnt)
  __builtin_amdgcn_s_wait_asynccnt(0);
#else
  asm volatile("s_wait_asynccnt 0" ::: "memory");
#endif
#else
  for (int c = threadIdx.x; c < nChunks; c += blockDim.x)
    ((uint4*)smemB)[c] = ((const uint4*)wB)[c];
#endif
  __syncthreads();
}

// ---------------------------------------------------------------------------
// Pack HWIO f32 weights into the V_WMMA_F32_16X16X32_F16 B-fragment layout.
// B (32x16, KxN) per wave32: lane L holds column n=L&15; half h=L>>4, VGPR v
// packs k=16*(v>>2)+8h+2*(v&3)+{0,1} (matches A table order).
// dst layout: [ntile][kstep][lane][8 dwords] (f16 pairs), 16 f16 per lane-step.
// ---------------------------------------------------------------------------
__global__ void pack_B(const float* __restrict__ w, _Float16* __restrict__ dst,
                       int Cout, int Ktotal, int Ksteps, int Ntiles) {
  int idx = blockIdx.x * blockDim.x + threadIdx.x;
  int total = Ntiles * Ksteps * 32 * 16;
  if (idx >= total) return;
  int pair  = idx & 1;
  int v     = (idx >> 1) & 7;
  int lane  = (idx >> 4) & 31;
  int rem   = idx >> 9;              // ntile*Ksteps + kstep
  int kstep = rem % Ksteps;
  int ntile = rem / Ksteps;
  int n = ntile * 16 + (lane & 15);
  int k = kstep * 32 + ((v >> 2) << 4) + ((lane >> 4) << 3) + ((v & 3) << 1) + pair;
  float val = 0.f;
  if (n < Cout && k < Ktotal) val = w[(size_t)k * Cout + n];  // HWIO: k = tap*Cin+ci
  dst[idx] = (_Float16)val;
}

__global__ void f32_to_f16(const float* __restrict__ src, _Float16* __restrict__ dst, int n) {
  int i = blockIdx.x * blockDim.x + threadIdx.x;
  if (i < n) dst[i] = (_Float16)src[i];
}

// ---------------------------------------------------------------------------
// ek0: 3x3 conv 1->32, stride 1, SAME, LeakyReLU. Direct (K=9 too small for WMMA).
// ---------------------------------------------------------------------------
__global__ void conv_e0(const float* __restrict__ in, const float* __restrict__ w,
                        const float* __restrict__ b, _Float16* __restrict__ out) {
  __shared__ float sw[288];
  __shared__ float sb[32];
  for (int i = threadIdx.x; i < 288; i += blockDim.x) sw[i] = w[i];
  for (int i = threadIdx.x; i < 32; i += blockDim.x) sb[i] = b[i];
  __syncthreads();
  int p = blockIdx.x * blockDim.x + threadIdx.x;
  if (p >= 16 * 128 * 128) return;
  int x = p & 127, y = (p >> 7) & 127, bb = p >> 14;
  float t[9];
#pragma unroll
  for (int ky = 0; ky < 3; ky++)
#pragma unroll
    for (int kx = 0; kx < 3; kx++) {
      int iy = y + ky - 1, ix = x + kx - 1;
      t[ky * 3 + kx] = (iy >= 0 && iy < 128 && ix >= 0 && ix < 128)
                           ? in[(bb << 14) + (iy << 7) + ix] : 0.f;
    }
  for (int c = 0; c < 32; c++) {
    float a = sb[c];
#pragma unroll
    for (int j = 0; j < 9; j++) a += t[j] * sw[j * 32 + c];
    out[(size_t)p * 32 + c] = (_Float16)lrelu(a);
  }
}

// ---------------------------------------------------------------------------
// Generic implicit-GEMM 3x3 conv via WMMA. One wave -> 16 output pixels (M tile)
// x NTILES*16 output channels. K = 9*Cin, stepped by 32 (one tap per step since
// 32 | Cin). Packed B weights are staged once into LDS (shared by all 4 waves,
// async LDS-DMA path) and read per K-step via ds_load_b128. A fragments are
// gathered from global NHWC f16 (two b128 per lane per step).
// ---------------------------------------------------------------------------
template <int NTILES>
__global__ __launch_bounds__(128)
void wmma_conv(const _Float16* __restrict__ in,   // [B,Hin,Win,Cin] f16
               const _Float16* __restrict__ wB,   // packed B fragments (global)
               const float* __restrict__ bias,    // [Cout]
               _Float16* __restrict__ outH,       // [B,Hout,Wout,Cout]
               float* __restrict__ outF,          // optional f32 dup
               const _Float16* __restrict__ res,  // optional post-lrelu add
               int Hin, int Win, int Cin,
               int Hout, int Wout,
               int Ksteps, int stride, int pad, int upsample) {
  extern __shared__ char smemRaw[];
  _Float16* smemB = (_Float16*)smemRaw;
  // 16-byte chunks: NTILES * Ksteps * 32 lanes * 16 f16 / 8 f16-per-chunk
  stage_B_to_lds(wB, smemB, NTILES * Ksteps * 64);

  const int Cout = NTILES * 16;
  const int lane = threadIdx.x & 31;
  const int wave = threadIdx.x >> 5;
  const int tile = blockIdx.x * (blockDim.x >> 5) + wave;
  const int half = lane >> 4;

  int pA = tile * 16 + (lane & 15);
  int xA = pA % Wout;
  int yA = (pA / Wout) % Hout;
  int bA = pA / (Wout * Hout);
  int HinE = upsample ? Hin * 2 : Hin;
  int WinE = upsample ? Win * 2 : Win;

  v8f zero = {};
  v8f acc[NTILES];
#pragma unroll
  for (int n = 0; n < NTILES; n++) acc[n] = zero;

  union AF { v16h h; unsigned u[8]; };
  union BF { v16h h; uint4 q[2]; };

  for (int ks = 0; ks < Ksteps; ks++) {
    int kbase = ks * 32;
    int tap = kbase / Cin;
    int ci0 = kbase - tap * Cin;
    int ky = tap / 3, kx = tap - ky * 3;
    int iy = yA * stride + ky - pad;
    int ix = xA * stride + kx - pad;
    bool ok = (iy >= 0 && iy < HinE && ix >= 0 && ix < WinE);
    if (upsample) { iy >>= 1; ix >>= 1; }
    size_t base = ((size_t)(bA * Hin + iy) * Win + ix) * (size_t)Cin + ci0;
    AF a;
#pragma unroll
    for (int v = 0; v < 8; v++) {
      int kk = ((v >> 2) << 4) + (half << 3) + ((v & 3) << 1);
      a.u[v] = ok ? *(const unsigned*)(in + base + kk) : 0u;
    }
#pragma unroll
    for (int n = 0; n < NTILES; n++) {
      BF bf;
      const uint4* bp = (const uint4*)(smemB + (size_t)((n * Ksteps + ks) * 32 + lane) * 16);
      bf.q[0] = bp[0];
      bf.q[1] = bp[1];
      acc[n] = __builtin_amdgcn_wmma_f32_16x16x32_f16(false, a.h, false, bf.h,
                                                      (short)0, acc[n], false, false);
    }
  }

  // Epilogue. C/D layout: lane L, VGPR r -> m = r + 8*(L>>4), n = L&15.
#pragma unroll
  for (int n = 0; n < NTILES; n++) {
    int col = n * 16 + (lane & 15);
    float bv = bias[col];
#pragma unroll
    for (int r = 0; r < 8; r++) {
      int m = r + (half << 3);
      size_t pix = (size_t)tile * 16 + m;
      size_t o = pix * Cout + col;
      float v = lrelu(acc[n][r] + bv);
      if (res) v += (float)res[o];
      outH[o] = (_Float16)v;
      if (outF) outF[o] = v;
    }
  }
}

// ---------------------------------------------------------------------------
// dyn_bias (1x1 conv 32->1) fused with img = inputs + lrelu(.)
// ---------------------------------------------------------------------------
__global__ void dyn_bias_img(const _Float16* __restrict__ d2, const float* __restrict__ wb,
                             const float* __restrict__ bb, const float* __restrict__ in0,
                             float* __restrict__ img) {
  __shared__ float sw[32];
  if (threadIdx.x < 32) sw[threadIdx.x] = wb[threadIdx.x];
  __syncthreads();
  int p = blockIdx.x * blockDim.x + threadIdx.x;
  if (p >= 16 * 128 * 128) return;
  float a = bb[0];
  const _Float16* d = d2 + (size_t)p * 32;
#pragma unroll
  for (int c = 0; c < 32; c++) a += (float)d[c] * sw[c];
  img[p] = in0[p] + lrelu(a);
}

// ---------------------------------------------------------------------------
// Final stage: dfk (1x1 conv 32->81, padded N=96 = 6 WMMA N-tiles, single
// K-step of 32) fused with the 9x9 dynamic filter application.
// Each lane applies its 6 filter taps per output row, then 16-lane shfl_xor
// reduction combines the N dimension.
// ---------------------------------------------------------------------------
__global__ __launch_bounds__(128)
void dyn_filter_apply(const _Float16* __restrict__ d2,
                      const _Float16* __restrict__ wB,   // packed dfk
                      const float* __restrict__ dfb,     // [81]
                      const float* __restrict__ img,     // [B,128,128]
                      float* __restrict__ out) {         // [B,128,128]
  const int lane = threadIdx.x & 31;
  const int wave = threadIdx.x >> 5;
  const int tile = blockIdx.x * (blockDim.x >> 5) + wave;
  const int half = lane >> 4;
  const int col = lane & 15;

  int pA = tile * 16 + col;
  union AF { v16h h; unsigned u[8]; };
  union BF { v16h h; uint4 q[2]; };
  AF a;
  const _Float16* src = d2 + (size_t)pA * 32;
#pragma unroll
  for (int v = 0; v < 8; v++) {
    int kk = ((v >> 2) << 4) + (half << 3) + ((v & 3) << 1);
    a.u[v] = *(const unsigned*)(src + kk);
  }
  v8f zero = {};
  v8f acc[6];
#pragma unroll
  for (int n = 0; n < 6; n++) {
    BF bf;
    const uint4* bp = (const uint4*)(wB + (size_t)(n * 32 + lane) * 16);
    bf.q[0] = bp[0];
    bf.q[1] = bp[1];
    acc[n] = __builtin_amdgcn_wmma_f32_16x16x32_f16(false, a.h, false, bf.h,
                                                    (short)0, zero, false, false);
  }
#pragma unroll
  for (int r = 0; r < 8; r++) {
    int m = r + (half << 3);
    int p = tile * 16 + m;
    int x = p & 127, y = (p >> 7) & 127, b = p >> 14;
    float partial = 0.f;
#pragma unroll
    for (int n = 0; n < 6; n++) {
      int t = n * 16 + col;
      if (t < 81) {
        float f = lrelu(acc[n][r] + dfb[t]);
        int ky = t / 9, kx = t - ky * 9;
        int iy = y + ky - 4, ix = x + kx - 4;
        float v = (iy >= 0 && iy < 128 && ix >= 0 && ix < 128)
                      ? img[(b << 14) + (iy << 7) + ix] : 0.f;
        partial += f * v;
      }
    }
    for (int off = 1; off < 16; off <<= 1)
      partial += __shfl_xor(partial, off, 32);
    if (col == 0) out[p] = partial;
  }
}

// ---------------------------------------------------------------------------
// Host launcher
// ---------------------------------------------------------------------------
extern "C" void kernel_launch(void* const* d_in, const int* in_sizes, int n_in,
                              void* d_out, int out_size, void* d_ws, size_t ws_size,
                              hipStream_t stream) {
  const float* inputs = (const float*)d_in[0];   // [16,128,128,1]
  const float* states = (const float*)d_in[1];   // [16,32,32,64]
  const float* ek0 = (const float*)d_in[2];  const float* eb0 = (const float*)d_in[3];
  const float* ek1 = (const float*)d_in[4];  const float* eb1 = (const float*)d_in[5];
  const float* ek2 = (const float*)d_in[6];  const float* eb2 = (const float*)d_in[7];
  const float* mk0 = (const float*)d_in[8];  const float* mb0 = (const float*)d_in[9];
  const float* mk1 = (const float*)d_in[10]; const float* mb1 = (const float*)d_in[11];
  const float* dk0 = (const float*)d_in[12]; const float* db0 = (const float*)d_in[13];
  const float* dk1 = (const float*)d_in[14]; const float* db1 = (const float*)d_in[15];
  const float* dk2 = (const float*)d_in[16]; const float* db2 = (const float*)d_in[17];
  const float* dfk = (const float*)d_in[18]; const float* dfb = (const float*)d_in[19];
  const float* dbk = (const float*)d_in[20]; const float* dbb = (const float*)d_in[21];

  char* ws = (char*)d_ws;
  size_t off = 0;
  auto alloc = [&](size_t bytes) {
    void* p = ws + off;
    off += (bytes + 255) & ~(size_t)255;
    return p;
  };
  _Float16* e0  = (_Float16*)alloc((size_t)16 * 128 * 128 * 32 * 2);  // 16.8 MB
  _Float16* e1  = (_Float16*)alloc((size_t)16 * 64 * 64 * 64 * 2);    //  8.4 MB
  _Float16* e2  = (_Float16*)alloc((size_t)16 * 32 * 32 * 64 * 2);    //  2 MB (h)
  _Float16* s16 = (_Float16*)alloc((size_t)16 * 32 * 32 * 64 * 2);
  _Float16* m0  = (_Float16*)alloc((size_t)16 * 32 * 32 * 64 * 2);
  _Float16* mid = (_Float16*)alloc((size_t)16 * 32 * 32 * 64 * 2);
  _Float16* d2b = (_Float16*)alloc((size_t)16 * 128 * 128 * 32 * 2);
  float*    img = (float*)alloc((size_t)16 * 128 * 128 * 4);
  _Float16* pwE1 = (_Float16*)alloc(4 * 9 * 32 * 16 * 2);
  _Float16* pwE2 = (_Float16*)alloc(4 * 18 * 32 * 16 * 2);
  _Float16* pwM0 = (_Float16*)alloc(4 * 18 * 32 * 16 * 2);
  _Float16* pwM1 = (_Float16*)alloc(4 * 18 * 32 * 16 * 2);
  _Float16* pwD0 = (_Float16*)alloc(4 * 18 * 32 * 16 * 2);
  _Float16* pwD1 = (_Float16*)alloc(2 * 18 * 32 * 16 * 2);
  _Float16* pwD2 = (_Float16*)alloc(2 * 9 * 32 * 16 * 2);
  _Float16* pwDF = (_Float16*)alloc(6 * 1 * 32 * 16 * 2);
  // Reuse dead buffers for decoder activations:
  _Float16* d0b = e1;   // e1 dead after ek2
  _Float16* d1b = e0;   // e0 dead after ek1

  float* out_img   = (float*)d_out;            // [16,128,128,1] = 262144
  float* new_state = (float*)d_out + 262144;   // [16,32,32,64]  = 1048576

  auto pack = [&](const float* w, _Float16* dst, int Cout, int Ktotal, int Ntiles) {
    int Ksteps = Ktotal / 32;
    int total = Ntiles * Ksteps * 32 * 16;
    hipLaunchKernelGGL(pack_B, dim3((total + 255) / 256), dim3(256), 0, stream,
                       w, dst, Cout, Ktotal, Ksteps, Ntiles);
  };
  pack(ek1, pwE1, 64, 288, 4);
  pack(ek2, pwE2, 64, 576, 4);
  pack(mk0, pwM0, 64, 576, 4);
  pack(mk1, pwM1, 64, 576, 4);
  pack(dk0, pwD0, 64, 576, 4);
  pack(dk1, pwD1, 32, 576, 2);
  pack(dk2, pwD2, 32, 288, 2);
  pack(dfk, pwDF, 81, 32, 6);

  hipLaunchKernelGGL(f32_to_f16, dim3(4096), dim3(256), 0, stream,
                     states, s16, 16 * 32 * 32 * 64);

  // Dynamic LDS bytes for packed B: NTILES * Ksteps * 32 lanes * 16 f16 * 2B
  auto ldsB = [](int Ntiles, int Ksteps) { return (size_t)Ntiles * Ksteps * 1024; };

  // encoder
  hipLaunchKernelGGL(conv_e0, dim3(1024), dim3(256), 0, stream, inputs, ek0, eb0, e0);
  // ek1: [16,128,128,32] -> [16,64,64,64], stride 2 (pad_lo=0)
  hipLaunchKernelGGL((wmma_conv<4>), dim3(1024), dim3(128), ldsB(4, 9), stream,
                     e0, pwE1, eb1, e1, (float*)nullptr, (const _Float16*)nullptr,
                     128, 128, 32, 64, 64, 9, 2, 0, 0);
  // ek2: [16,64,64,64] -> [16,32,32,64], stride 2
  hipLaunchKernelGGL((wmma_conv<4>), dim3(256), dim3(128), ldsB(4, 18), stream,
                     e1, pwE2, eb2, e2, (float*)nullptr, (const _Float16*)nullptr,
                     64, 64, 64, 32, 32, 18, 2, 0, 0);
  // mid
  hipLaunchKernelGGL((wmma_conv<4>), dim3(256), dim3(128), ldsB(4, 18), stream,
                     s16, pwM0, mb0, m0, (float*)nullptr, (const _Float16*)nullptr,
                     32, 32, 64, 32, 32, 18, 1, 1, 0);
  // mk1 + residual(h) -> mid (f16) and new_state (f32)
  hipLaunchKernelGGL((wmma_conv<4>), dim3(256), dim3(128), ldsB(4, 18), stream,
                     m0, pwM1, mb1, mid, new_state, e2,
                     32, 32, 64, 32, 32, 18, 1, 1, 0);
  // decoder
  // dk0: up2 + conv 64->64 -> [16,64,64,64]
  hipLaunchKernelGGL((wmma_conv<4>), dim3(1024), dim3(128), ldsB(4, 18), stream,
                     mid, pwD0, db0, d0b, (float*)nullptr, (const _Float16*)nullptr,
                     32, 32, 64, 64, 64, 18, 1, 1, 1);
  // dk1: up2 + conv 64->32 -> [16,128,128,32]
  hipLaunchKernelGGL((wmma_conv<2>), dim3(4096), dim3(128), ldsB(2, 18), stream,
                     d0b, pwD1, db1, d1b, (float*)nullptr, (const _Float16*)nullptr,
                     64, 64, 64, 128, 128, 18, 1, 1, 1);
  // dk2: conv 32->32 -> [16,128,128,32]
  hipLaunchKernelGGL((wmma_conv<2>), dim3(4096), dim3(128), ldsB(2, 9), stream,
                     d1b, pwD2, db2, d2b, (float*)nullptr, (const _Float16*)nullptr,
                     128, 128, 32, 128, 128, 9, 1, 1, 0);

  // dynamic bias -> img, then fused dfk WMMA + 9x9 filter application
  hipLaunchKernelGGL(dyn_bias_img, dim3(1024), dim3(256), 0, stream,
                     d2b, dbk, dbb, inputs, img);
  hipLaunchKernelGGL(dyn_filter_apply, dim3(4096), dim3(128), 0, stream,
                     d2b, pwDF, dfb, img, out_img);
  (void)in_sizes; (void)n_in; (void)out_size; (void)ws_size;
}
